// HealPIXAttention_19404662243884
// MI455X (gfx1250) — matrix operand order repaired and checked by reading, not compiled, not run on hardware
//
#include <hip/hip_runtime.h>
#include <hip/hip_bf16.h>

// ---------------------------------------------------------------------------
// MI455X (gfx1250): GroupNorm + 8-head attention + proj.
// GEMMs on v_wmma_f32_16x16x32_f16; global->LDS staging uses the CDNA5
// async-to-LDS path (ASYNCcnt) when the builtin is available.
// ---------------------------------------------------------------------------

typedef _Float16 h16;
typedef __attribute__((ext_vector_type(16))) _Float16 v16h;
typedef __attribute__((ext_vector_type(8)))  _Float16 v8h;
typedef __attribute__((ext_vector_type(8)))  float    v8f;
typedef __attribute__((ext_vector_type(4)))  int      v4i;

union FragAB { v16h v; v8h h2[2]; };

static constexpr int kC   = 512;    // channels
static constexpr int kN   = 3072;   // pixels
static constexpr int kH   = 8;      // heads
static constexpr int kCPH = 64;     // channels per head
static constexpr int kG   = 32;     // groups

__device__ __forceinline__ v8f wmma16(const FragAB& a, const FragAB& b, v8f c) {
  return __builtin_amdgcn_wmma_f32_16x16x32_f16(
      /*neg_a=*/false, a.v, /*neg_b=*/false, b.v,
      /*c_mod=*/(short)0, c, /*reuse_a=*/false, /*reuse_b=*/false);
}

// 16B global -> LDS copy, async (ASYNCcnt) when available.
__device__ __forceinline__ void cp16(const h16* g, h16* l) {
#if __has_builtin(__builtin_amdgcn_global_load_async_to_lds_b128)
  __builtin_amdgcn_global_load_async_to_lds_b128(
      (__attribute__((address_space(1))) v4i*)g,
      (__attribute__((address_space(3))) v4i*)l, 0, 0);
#else
  *(v8h*)l = *(const v8h*)g;
#endif
}

__device__ __forceinline__ void cp_wait() {
#if __has_builtin(__builtin_amdgcn_global_load_async_to_lds_b128)
#if __has_builtin(__builtin_amdgcn_s_wait_asynccnt)
  __builtin_amdgcn_s_wait_asynccnt(0);
#else
  asm volatile("s_wait_asynccnt 0" ::: "memory");
#endif
#endif
}

// ---------------------------------------------------------------------------
// f32 -> f16 convert
// ---------------------------------------------------------------------------
__global__ void cvt_f16_k(const float* __restrict__ s, h16* __restrict__ d, int n) {
  int i = blockIdx.x * blockDim.x + threadIdx.x;
  if (i < n) d[i] = (h16)s[i];
}

// ---------------------------------------------------------------------------
// GroupNorm: one block per group of 16 channels; writes xn transposed [n][c].
// ---------------------------------------------------------------------------
__global__ __launch_bounds__(256) void groupnorm_k(
    const float* __restrict__ x, const float* __restrict__ w,
    const float* __restrict__ b, h16* __restrict__ xn_t) {
  __shared__ float ss[256], sq[256];
  const int g = blockIdx.x, t = threadIdx.x;
  const int c0 = g * (kC / kG);
  const int total = (kC / kG) * kN;             // 49152
  float s = 0.f, q = 0.f;
  for (int i = t; i < total; i += 256) {
    float v = x[(c0 + i / kN) * kN + (i % kN)];
    s += v; q += v * v;
  }
  ss[t] = s; sq[t] = q; __syncthreads();
  for (int k = 128; k > 0; k >>= 1) {
    if (t < k) { ss[t] += ss[t + k]; sq[t] += sq[t + k]; }
    __syncthreads();
  }
  const float mean = ss[0] / (float)total;
  const float var  = sq[0] / (float)total - mean * mean;
  const float rinv = rsqrtf(var + 1e-5f);
  for (int i = t; i < total; i += 256) {
    int c = c0 + i / kN, n = i % kN;
    float xn = (x[c * kN + n] - mean) * rinv * w[c] + b[c];
    xn_t[n * kC + c] = (h16)xn;
  }
}

// ---------------------------------------------------------------------------
// QKV GEMM: (1536x512) x (512x3072). Block tile 128(M) x 64(N), K-step 64,
// 8 waves; 8 WMMAs per sync window. Epilogue de-interleaves q/k/v.
// ---------------------------------------------------------------------------
__global__ __launch_bounds__(256) void qkv_gemm_k(
    const h16* __restrict__ A,      // qkv_w f16 [1536][512]
    const h16* __restrict__ Bt,     // xn_t     [3072][512]
    const float* __restrict__ bias,
    h16* __restrict__ q_t, h16* __restrict__ k_t, h16* __restrict__ v_n) {
  __shared__ h16 As[128 * 64];
  __shared__ h16 Bs[64 * 64];
  const int t = threadIdx.x, wave = t >> 5, lane = t & 31;
  const int Mb = blockIdx.y * 128, Nb = blockIdx.x * 64;
  v8f acc[4] = {};

  for (int kk = 0; kk < kC; kk += 64) {
    for (int i = t; i < 128 * 8; i += 256) {       // 1024 x 16B chunks
      int row = i >> 3, c8 = (i & 7) * 8;
      cp16(&A[(Mb + row) * kC + kk + c8], &As[row * 64 + c8]);
    }
    for (int i = t; i < 64 * 8; i += 256) {        // 512 x 16B chunks
      int row = i >> 3, c8 = (i & 7) * 8;
      cp16(&Bt[(Nb + row) * kC + kk + c8], &Bs[row * 64 + c8]);
    }
    cp_wait();
    __syncthreads();

    FragAB af0, af1;   // A frag: M = lane%16, K = (lane/16)*16 + h
    const int ar = (wave * 16 + (lane & 15)) * 64 + (lane >> 4) * 16;
    af0.h2[0] = *(const v8h*)&As[ar];
    af0.h2[1] = *(const v8h*)&As[ar + 8];
    af1.h2[0] = *(const v8h*)&As[ar + 32];
    af1.h2[1] = *(const v8h*)&As[ar + 40];
#pragma unroll
    for (int nb = 0; nb < 4; nb++) {
      FragAB bf; // B frag: N = lane%16, K = (lane/16)*16 + h  (Bs is [n][k])
      const int br = (nb * 16 + (lane & 15)) * 64 + (lane >> 4) * 16;
      bf.v = *(const v16h*)&Bs[br];
      acc[nb] = wmma16(af0, bf, acc[nb]);
      bf.v = *(const v16h*)&Bs[br + 32];
      acc[nb] = wmma16(af1, bf, acc[nb]);
    }
    __syncthreads();
  }

#pragma unroll
  for (int nb = 0; nb < 4; nb++) {
    int n = Nb + nb * 16 + (lane & 15);
#pragma unroll
    for (int r = 0; r < 8; r++) {
      int o = Mb + wave * 16 + (lane >> 4) * 8 + r;   // D: M = r + 8*(lane/16)
      float val = acc[nb][r] + bias[o];
      int s = o % 3, c = (o / 3) & (kCPH - 1), hh = o / (3 * kCPH);
      if (s == 0)      q_t[(hh * kN + n) * kCPH + c] = (h16)(val * 0.125f);
      else if (s == 1) k_t[(hh * kN + n) * kCPH + c] = (h16)val;
      else             v_n[(hh * kCPH + c) * kN + n] = (h16)val;
    }
  }
}

// ---------------------------------------------------------------------------
// Flash attention: grid (48 q-tiles, 8 heads), 128 threads = 4 waves,
// each wave owns 16 queries; streams K/V in 32-wide chunks.
// ---------------------------------------------------------------------------
__global__ __launch_bounds__(128) void attn_k(
    const h16* __restrict__ q_t,   // [h][n][c]
    const h16* __restrict__ k_t,   // [h][n][c]
    const h16* __restrict__ v_n,   // [h][c][n]
    h16* __restrict__ a_t) {       // out [n][C] transposed
  __shared__ h16 Qs[64 * 64];
  __shared__ h16 Ks[32 * 64];
  __shared__ h16 Vs[64 * 32];
  __shared__ h16 Ps[4][16 * 32];
  __shared__ float st[4][16];
  const int t = threadIdx.x, wave = t >> 5, lane = t & 31;
  const int qb = blockIdx.x * 64, hh = blockIdx.y;

  for (int i = t; i < 64 * 8; i += 128) {
    int row = i >> 3, c8 = (i & 7) * 8;
    cp16(&q_t[(hh * kN + qb + row) * kCPH + c8], &Qs[row * 64 + c8]);
  }
  cp_wait();
  __syncthreads();

  FragAB aq[2];  // A = Q^T tile: M = query, Kdim = c (two 32-c halves)
#pragma unroll
  for (int j = 0; j < 2; j++) {
    aq[j].h2[0] = *(const v8h*)&Qs[(wave * 16 + (lane & 15)) * 64 + j * 32 + (lane >> 4) * 16];
    aq[j].h2[1] = *(const v8h*)&Qs[(wave * 16 + (lane & 15)) * 64 + j * 32 + (lane >> 4) * 16 + 8];
  }

  float m[8], l[8];
#pragma unroll
  for (int r = 0; r < 8; r++) { m[r] = -3.0e38f; l[r] = 0.f; }
  v8f O[4] = {};

  for (int kb = 0; kb < kN; kb += 32) {
    __syncthreads();   // previous chunk's Ks/Vs reads complete
    for (int i = t; i < 32 * 8; i += 128) {
      int row = i >> 3, c8 = (i & 7) * 8;
      cp16(&k_t[(hh * kN + kb + row) * kCPH + c8], &Ks[row * 64 + c8]);
    }
    for (int i = t; i < 64 * 4; i += 128) {
      int row = i >> 2, c8 = (i & 3) * 8;
      cp16(&v_n[(hh * kCPH + row) * kN + kb + c8], &Vs[row * 32 + c8]);
    }
    cp_wait();
    __syncthreads();

    // S = Q^T K  (16 queries x 32 keys), reduce over c = 64 (2 WMMAs each)
    v8f S0 = {}, S1 = {};
    {
      FragAB bf;
      bf.v = *(const v16h*)&Ks[((lane & 15)) * 64 + (lane >> 4) * 16];
      S0 = wmma16(aq[0], bf, S0);
      bf.v = *(const v16h*)&Ks[((lane & 15)) * 64 + 32 + (lane >> 4) * 16];
      S0 = wmma16(aq[1], bf, S0);
      bf.v = *(const v16h*)&Ks[(16 + (lane & 15)) * 64 + (lane >> 4) * 16];
      S1 = wmma16(aq[0], bf, S1);
      bf.v = *(const v16h*)&Ks[(16 + (lane & 15)) * 64 + 32 + (lane >> 4) * 16];
      S1 = wmma16(aq[1], bf, S1);
    }

    // online softmax: each row lives across a 16-lane half (xor 8..1 stays in)
    float p0[8], p1[8], alpha[8];
#pragma unroll
    for (int r = 0; r < 8; r++) {
      float cm = fmaxf(S0[r], S1[r]);
#pragma unroll
      for (int d = 8; d >= 1; d >>= 1) cm = fmaxf(cm, __shfl_xor(cm, d, 32));
      float mn = fmaxf(m[r], cm);
      alpha[r] = __expf(m[r] - mn);
      p0[r] = __expf(S0[r] - mn);
      p1[r] = __expf(S1[r] - mn);
      float rs = p0[r] + p1[r];
#pragma unroll
      for (int d = 8; d >= 1; d >>= 1) rs += __shfl_xor(rs, d, 32);
      l[r] = l[r] * alpha[r] + rs;
      m[r] = mn;
    }

    // P -> LDS in B-fragment layout [q][k] (wave-private region)
#pragma unroll
    for (int r = 0; r < 8; r++) {
      int q = (lane >> 4) * 8 + r;
      Ps[wave][q * 32 + (lane & 15)]      = (h16)p0[r];
      Ps[wave][q * 32 + 16 + (lane & 15)] = (h16)p1[r];
    }
    if ((lane & 15) == 0) {
#pragma unroll
      for (int r = 0; r < 8; r++) st[wave][(lane >> 4) * 8 + r] = alpha[r];
    }
    float aQ = st[wave][lane & 15];   // per-lane column rescale factor

    // O[c,q] = alpha*O + V(64c x 32k) * P^T(32k x 16q)
#pragma unroll
    for (int cb = 0; cb < 4; cb++) {
      FragAB av, bp;
      av.h2[0] = *(const v8h*)&Vs[(cb * 16 + (lane & 15)) * 32 + (lane >> 4) * 16];
      av.h2[1] = *(const v8h*)&Vs[(cb * 16 + (lane & 15)) * 32 + (lane >> 4) * 16 + 8];
      bp.v = *(const v16h*)&Ps[wave][(lane & 15) * 32 + (lane >> 4) * 16];
      O[cb] = O[cb] * aQ;
      O[cb] = wmma16(av, bp, O[cb]);
    }
  }

  if ((lane & 15) == 0) {
#pragma unroll
    for (int r = 0; r < 8; r++) st[wave][(lane >> 4) * 8 + r] = 1.0f / l[r];
  }
  float li = st[wave][lane & 15];
#pragma unroll
  for (int cb = 0; cb < 4; cb++) {
    v8h ov;
#pragma unroll
    for (int r = 0; r < 8; r++) ov[r] = (h16)(O[cb][r] * li);
    int q = qb + wave * 16 + (lane & 15);
    *(v8h*)&a_t[q * kC + hh * kCPH + cb * 16 + (lane >> 4) * 8] = ov;
  }
}

// ---------------------------------------------------------------------------
// Proj GEMM + bias + residual: out = x + proj_w @ a
// ---------------------------------------------------------------------------
__global__ __launch_bounds__(256) void proj_gemm_k(
    const h16* __restrict__ A,      // proj_w f16 [512][512]
    const h16* __restrict__ Bt,     // a_t        [3072][512]
    const float* __restrict__ bias, const float* __restrict__ x,
    float* __restrict__ out) {
  __shared__ h16 As[128 * 64];
  __shared__ h16 Bs[64 * 64];
  const int t = threadIdx.x, wave = t >> 5, lane = t & 31;
  const int Mb = blockIdx.y * 128, Nb = blockIdx.x * 64;
  v8f acc[4] = {};

  for (int kk = 0; kk < kC; kk += 64) {
    for (int i = t; i < 128 * 8; i += 256) {
      int row = i >> 3, c8 = (i & 7) * 8;
      cp16(&A[(Mb + row) * kC + kk + c8], &As[row * 64 + c8]);
    }
    for (int i = t; i < 64 * 8; i += 256) {
      int row = i >> 3, c8 = (i & 7) * 8;
      cp16(&Bt[(Nb + row) * kC + kk + c8], &Bs[row * 64 + c8]);
    }
    cp_wait();
    __syncthreads();

    FragAB af0, af1;
    const int ar = (wave * 16 + (lane & 15)) * 64 + (lane >> 4) * 16;
    af0.h2[0] = *(const v8h*)&As[ar];
    af0.h2[1] = *(const v8h*)&As[ar + 8];
    af1.h2[0] = *(const v8h*)&As[ar + 32];
    af1.h2[1] = *(const v8h*)&As[ar + 40];
#pragma unroll
    for (int nb = 0; nb < 4; nb++) {
      FragAB bf;
      const int br = (nb * 16 + (lane & 15)) * 64 + (lane >> 4) * 16;
      bf.v = *(const v16h*)&Bs[br];
      acc[nb] = wmma16(af0, bf, acc[nb]);
      bf.v = *(const v16h*)&Bs[br + 32];
      acc[nb] = wmma16(af1, bf, acc[nb]);
    }
    __syncthreads();
  }

#pragma unroll
  for (int nb = 0; nb < 4; nb++) {
    int n = Nb + nb * 16 + (lane & 15);
#pragma unroll
    for (int r = 0; r < 8; r++) {
      int o = Mb + wave * 16 + (lane >> 4) * 8 + r;
      out[o * kN + n] = acc[nb][r] + bias[o] + x[o * kN + n];
    }
  }
}

// ---------------------------------------------------------------------------
// Launch. Workspace: f16 buffers, 256B aligned, ~18 MB total.
// ---------------------------------------------------------------------------
extern "C" void kernel_launch(void* const* d_in, const int* in_sizes, int n_in,
                              void* d_out, int out_size, void* d_ws, size_t ws_size,
                              hipStream_t stream) {
  const float* x      = (const float*)d_in[0];
  const float* norm_w = (const float*)d_in[1];
  const float* norm_b = (const float*)d_in[2];
  const float* qkv_w  = (const float*)d_in[3];
  const float* qkv_b  = (const float*)d_in[4];
  const float* proj_w = (const float*)d_in[5];
  const float* proj_b = (const float*)d_in[6];
  float* out = (float*)d_out;

  char* ws = (char*)d_ws;
  size_t off = 0;
  auto alloc_h16 = [&](size_t elems) -> h16* {
    h16* p = (h16*)(ws + off);
    off = (off + elems * sizeof(h16) + 255) & ~(size_t)255;
    return p;
  };
  h16* qkv_w16  = alloc_h16((size_t)3 * kC * kC);
  h16* proj_w16 = alloc_h16((size_t)kC * kC);
  h16* xn_t = alloc_h16((size_t)kN * kC);
  h16* q_t  = alloc_h16((size_t)kH * kN * kCPH);
  h16* k_t  = alloc_h16((size_t)kH * kN * kCPH);
  h16* v_n  = alloc_h16((size_t)kH * kCPH * kN);
  h16* a_t  = alloc_h16((size_t)kN * kC);

  const int nQW = 3 * kC * kC, nPW = kC * kC;
  cvt_f16_k<<<(nQW + 255) / 256, 256, 0, stream>>>(qkv_w, qkv_w16, nQW);
  cvt_f16_k<<<(nPW + 255) / 256, 256, 0, stream>>>(proj_w, proj_w16, nPW);

  groupnorm_k<<<kG, 256, 0, stream>>>(x, norm_w, norm_b, xn_t);

  qkv_gemm_k<<<dim3(kN / 64, (3 * kC) / 128), 256, 0, stream>>>(
      qkv_w16, xn_t, qkv_b, q_t, k_t, v_n);

  attn_k<<<dim3(kN / 64, kH), 128, 0, stream>>>(q_t, k_t, v_n, a_t);

  proj_gemm_k<<<dim3(kN / 64, kC / 128), 256, 0, stream>>>(
      proj_w16, a_t, proj_b, x, out);
}
